// LSTMSequentialEncoder_56865366999082
// MI455X (gfx1250) — matrix operand order repaired and checked by reading, not compiled, not run on hardware
//
#include <hip/hip_runtime.h>
#include <hip/hip_bf16.h>

// ---------------------------------------------------------------------------
// ConvLSTM (B=8,T=16,C=64,HD=64,H=W=64), 3x3 SAME gate conv, bias-free.
// Per timestep: implicit GEMM  conv[256, 32768] = Wpk[256,1152] x Z[1152,32768]
// in bf16 via v_wmma_f32_16x16x32_bf16 (f32 accumulate), fused LSTM gates.
// ---------------------------------------------------------------------------

typedef __attribute__((ext_vector_type(16))) __bf16        v16bf;
typedef __attribute__((ext_vector_type(8)))  float         v8f;
typedef __attribute__((ext_vector_type(4)))  unsigned int  u32x4;

union Frag16 {
    v16bf        bf;
    unsigned int u[8];
    u32x4        q[2];
};

#define B_    8
#define T_    16
#define CIN   64
#define HD_   64
#define H_    64
#define W_    64
#define ZC    128          // CIN + HD
#define OC_   256          // 4 * HD
#define KTOT  1152         // ZC * 9
#define NIMG  (H_ * W_)    // 4096
#define HIDE  (B_ * HD_ * H_ * W_)   // elements per hid/cell buffer = 2097152

// round-to-nearest-even f32 -> bf16 (bit exact, no reliance on __bf16 cvt)
__device__ __forceinline__ unsigned short f2bf(float f) {
    unsigned int u = __float_as_uint(f);
    unsigned int r = (u + 0x7FFFu + ((u >> 16) & 1u)) >> 16;
    return (unsigned short)r;
}

__device__ __forceinline__ float sigf(float v) {
    return 1.0f / (1.0f + __expf(-v));
}

// ---------------------------------------------------------------------------
// Weight prepack: Wk (256,128,3,3) f32  ->  wpk[oc][tap*128+ic] bf16
// ---------------------------------------------------------------------------
__global__ void prepack_w(const float* __restrict__ Wk,
                          unsigned short* __restrict__ wpk) {
    int oc = blockIdx.x;     // 256
    int ic = threadIdx.x;    // 128
#pragma unroll
    for (int tap = 0; tap < 9; ++tap) {
        int ky = tap / 3, kx = tap % 3;
        float v = Wk[((oc * ZC + ic) * 3 + ky) * 3 + kx];
        wpk[(size_t)oc * KTOT + tap * ZC + ic] = f2bf(v);
    }
}

// ---------------------------------------------------------------------------
// One ConvLSTM step. Block = 256 threads (8 wave32), grid = B_*H_ = 512.
// Block computes all 256 out-channels for one image row (64 pixels),
// then applies the LSTM gate update in place.
// ---------------------------------------------------------------------------
__global__ __launch_bounds__(256) void convlstm_step(
    const float* __restrict__ x, int t,
    const unsigned short* __restrict__ wpk,
    const unsigned short* __restrict__ hid_in,   // bf16
    unsigned short* __restrict__ hid_out,        // bf16
    const float* __restrict__ cell_in,
    float* __restrict__ cell_out,
    float* __restrict__ hid_f32_out)             // nullptr except last step
{
    // 64KB LDS union:
    //  phase 1: z tile, channel-pair interleaved bf16: [3 rows][64 cpair][66 col]
    //           word(r,cp,col) = { ch=2cp lo16, ch=2cp+1 hi16 }   (12672 words)
    //  phase 2: conv result f32 [256 oc][64 x]                     (16384 words)
    __shared__ unsigned int smem[16384];

    const int b = blockIdx.x >> 6;   // image
    const int y = blockIdx.x & 63;   // row

    // ---------------- stage z tile (with zero halo) into LDS ----------------
    {
        const float* xb = x + ((size_t)(b * T_ + t)) * CIN * NIMG;
        const unsigned short* hb = hid_in + (size_t)b * HD_ * NIMG;
        for (int i = threadIdx.x; i < 3 * 64 * 66; i += 256) {
            int col  = i % 66;          // xx = col - 1
            int rest = i / 66;
            int cp   = rest & 63;       // channel pair
            int r    = rest >> 6;       // 0..2  (yy = y + r - 1)
            int yy = y + r - 1;
            int xx = col - 1;
            unsigned int word = 0;
            if (yy >= 0 && yy < H_ && xx >= 0 && xx < W_) {
                int ch0 = cp * 2;
                unsigned short lo, hi;
                if (ch0 < CIN) {
                    lo = f2bf(xb[(ch0 * H_ + yy) * W_ + xx]);
                    hi = f2bf(xb[((ch0 + 1) * H_ + yy) * W_ + xx]);
                } else {
                    lo = hb[((ch0 - CIN) * H_ + yy) * W_ + xx];
                    hi = hb[((ch0 + 1 - CIN) * H_ + yy) * W_ + xx];
                }
                word = (unsigned int)lo | ((unsigned int)hi << 16);
            }
            smem[i] = word;
        }
    }
    __syncthreads();

    // ---------------- implicit GEMM: 36 K-chunks of 32 ----------------------
    const int lane = threadIdx.x & 31;
    const int wv   = threadIdx.x >> 5;   // wave 0..7 -> oc rows [32w,32w+32)
    const int mrow = lane & 15;
    const int sel  = lane >> 4;          // 0: lanes 0-15, 1: lanes 16-31

    v8f acc[2][4] = {};  // [oc subtile][n subtile], 16x16 f32 each

    const unsigned short* w0 = wpk + (size_t)(wv * 32 + mrow) * KTOT;

    for (int c = 0; c < 36; ++c) {
        const int kc0 = c * 32;                 // == tap*128 + ic0
        const int tap = c >> 2;
        const int ic0 = (c & 3) * 32;
        const int ky  = tap / 3;
        const int kx  = tap - ky * 3;

        // A fragments (weights, contiguous bf16 rows in global/L2)
        Frag16 a[2];
#pragma unroll
        for (int i = 0; i < 2; ++i) {
            const unsigned short* wrow = w0 + (size_t)i * 16 * KTOT + kc0;
            a[i].q[0] = *(const u32x4*)(wrow + sel * 16);       // K sel*16..+8
            a[i].q[1] = *(const u32x4*)(wrow + sel * 16 + 8);   // K sel*16+8..+8
        }
        if (c + 1 < 36)
            __builtin_prefetch((const void*)(w0 + kc0 + 32), 0, 3);

        // B fragments from LDS (channel pairs along K, column = x + kx)
        Frag16 bfrag[4];
        const int cpbase = (ic0 >> 1) + sel * 8;
#pragma unroll
        for (int j = 0; j < 4; ++j) {
            const int col = j * 16 + mrow + kx;           // 0..65
            const unsigned int* zp = smem + (ky * 64 + cpbase) * 66 + col;
#pragma unroll
            for (int v = 0; v < 8; ++v) bfrag[j].u[v] = zp[v * 66];
        }

#pragma unroll
        for (int i = 0; i < 2; ++i)
#pragma unroll
            for (int j = 0; j < 4; ++j)
                acc[i][j] = __builtin_amdgcn_wmma_f32_16x16x32_bf16(
                    /*neg_a=*/false, a[i].bf,
                    /*neg_b=*/false, bfrag[j].bf,
                    /*c_mod=*/(short)0, acc[i][j],
                    /*reuse_a=*/false, /*reuse_b=*/false);
    }

    // ---------------- spill conv results to LDS (reuse z buffer) ------------
    __syncthreads();
    float* conv = (float*)smem;   // [256 oc][64 x]
#pragma unroll
    for (int i = 0; i < 2; ++i) {
        const int mbase = wv * 32 + i * 16 + sel * 8;
#pragma unroll
        for (int j = 0; j < 4; ++j) {
            const int n = j * 16 + mrow;
#pragma unroll
            for (int r = 0; r < 8; ++r)
                conv[(mbase + r) * 64 + n] = acc[i][j][r];
        }
    }
    __syncthreads();

    // ---------------- fused LSTM gate update --------------------------------
    for (int it = threadIdx.x; it < HD_ * W_; it += 256) {
        const int hc = it >> 6;     // hidden channel
        const int xc = it & 63;     // column
        const float ci = conv[(hc      ) * 64 + xc];
        const float cf = conv[(hc +  64) * 64 + xc];
        const float co = conv[(hc + 128) * 64 + xc];
        const float cg = conv[(hc + 192) * 64 + xc];
        const float ig = sigf(ci);
        const float fg = sigf(cf);
        const float og = sigf(co);
        const float g  = tanhf(cg);
        const size_t gi = (((size_t)b * HD_ + hc) * H_ + y) * W_ + xc;
        const float cnew = fg * cell_in[gi] + ig * g;
        cell_out[gi] = cnew;
        const float hv = og * tanhf(cnew);
        hid_out[gi] = f2bf(hv);
        if (hid_f32_out) hid_f32_out[gi] = hv;
    }
}

// ---------------------------------------------------------------------------
// Host launcher (graph-capture safe: only async ops on `stream`).
// ws layout:
//   [0)              wpk   bf16  256*1152          =   589,824 B
//   [+589824)        hidA  bf16  HIDE              = 4,194,304 B
//   [+...)           hidB  bf16  HIDE              = 4,194,304 B
//   [+...)           cell  f32   HIDE              = 8,388,608 B
// total ~16.6 MB
// ---------------------------------------------------------------------------
extern "C" void kernel_launch(void* const* d_in, const int* in_sizes, int n_in,
                              void* d_out, int out_size, void* d_ws, size_t ws_size,
                              hipStream_t stream) {
    (void)in_sizes; (void)n_in; (void)out_size; (void)ws_size;

    const float* x  = (const float*)d_in[0];
    const float* Wk = (const float*)d_in[1];

    char* ws = (char*)d_ws;
    unsigned short* wpk  = (unsigned short*)ws;
    unsigned short* hidA = (unsigned short*)(ws + 589824);
    unsigned short* hidB = hidA + (size_t)HIDE;
    float*          cell = (float*)(hidB + (size_t)HIDE);

    float* outHid  = (float*)d_out;
    float* outCell = outHid + (size_t)HIDE;

    hipMemsetAsync(hidA, 0, (size_t)HIDE * sizeof(unsigned short), stream);
    hipMemsetAsync(cell, 0, (size_t)HIDE * sizeof(float), stream);

    prepack_w<<<dim3(OC_), dim3(ZC), 0, stream>>>(Wk, wpk);

    unsigned short* hin  = hidA;
    unsigned short* hout = hidB;
    for (int t = 0; t < T_; ++t) {
        const bool last = (t == T_ - 1);
        convlstm_step<<<dim3(B_ * H_), dim3(256), 0, stream>>>(
            x, t, wpk, hin, hout,
            cell, last ? outCell : cell,
            last ? outHid : nullptr);
        unsigned short* tmp = hin; hin = hout; hout = tmp;
    }
}
